// MambaBlockTVM_90452011253775
// MI455X (gfx1250) — compile-verified, matrix operand order checked by reading, not run on hardware
//
#include <hip/hip_runtime.h>
#include <hip/hip_bf16.h>
#include <math.h>

// ---------------------------------------------------------------------------
// Mamba block for MI455X (gfx1250, wave32, WMMA + TDM).
// B=8, L=4096, D_MODEL=768, D_INNER=1536, D_CONV=4, D_STATE=16, DT_RANK=48
// ---------------------------------------------------------------------------

#define BSZ    8
#define LSEQ   4096
#define DMODEL 768
#define DINNER 1536
#define DCONV  4
#define DSTATE 16
#define DTRANK 48
#define XPN    (DTRANK + 2 * DSTATE)   // 80

typedef __attribute__((ext_vector_type(16))) __bf16       v16bf;
typedef __attribute__((ext_vector_type(4)))  __bf16       v4bf;
typedef __attribute__((ext_vector_type(8)))  float        v8f;
typedef __attribute__((ext_vector_type(4)))  unsigned int u32x4;
typedef __attribute__((ext_vector_type(8)))  unsigned int u32x8;

#if defined(__HIP_DEVICE_COMPILE__) && defined(__has_builtin)
#if __has_builtin(__builtin_amdgcn_s_wait_tensorcnt)
#define USE_TDM 1
#endif
#endif

static __device__ inline v4bf cvt4(float4 v)
{
    v4bf r;
    r[0] = (__bf16)v.x; r[1] = (__bf16)v.y;
    r[2] = (__bf16)v.z; r[3] = (__bf16)v.w;
    return r;
}

// ---------------------------------------------------------------------------
// GEMM: C[m,n] = sum_k A[m,k] * W[n,k]   (both row-major, K contiguous)
//
// Block tile 128(M) x 128(N), K stepped by 32, DOUBLE-BUFFERED LDS (32 KB):
// next tile's global_load_b128s are issued before the WMMA block so HBM
// latency overlaps matrix math; one barrier per K-step.
// 256 threads = 8 waves in 4(M) x 2(N); each wave owns 32x64 = 2x4 wmma
// tiles -> 8 v_wmma per K-step, 128*128*32 MACs per 16 KB staged.
// Guards are whole-float4 predicates (requires K % 4 == 0; K = 768/1536/48).
// EPILOGUE: 0 = store; 1 = bias + softplus (fused dt_proj -> delta).
// ---------------------------------------------------------------------------
template <int EPILOGUE>
__global__ __launch_bounds__(256)
void gemm_bf16_wmma(const float* __restrict__ A, long lda,
                    const float* __restrict__ W,      // [N, K] row-major
                    const float* __restrict__ bias,   // [N] (EPILOGUE==1)
                    float* __restrict__ C, long ldc,
                    int M, int N, int K)
{
    constexpr int BM = 128, BN = 128, KT = 32;
    __shared__ __align__(32) __bf16 As[2][BM][KT];   // 16 KB
    __shared__ __align__(32) __bf16 Ws[2][BN][KT];   // 16 KB

    const int tid   = threadIdx.x;
    const int mBase = blockIdx.y * BM;
    const int nBase = blockIdx.x * BN;
    const int wave  = tid >> 5;
    const int lane  = tid & 31;
    const int wm    = wave >> 1;        // 0..3  (rows of 32)
    const int wn    = wave & 1;         // 0..1  (cols of 64)
    const int row16 = lane & 15;
    const int kh    = (lane >> 4) << 4; // 0 or 16 (K-half per lane group)

    // fixed per-thread staging coordinates: 4 float4 each for A and W
    int er[4], ec[4];
    #pragma unroll
    for (int i = 0; i < 4; ++i) {
        int e = tid * 4 + i * 1024;
        er[i] = e >> 5; ec[i] = e & 31;
    }

    float4 ra[4], rw[4];
    auto gload = [&](int k0) {
        #pragma unroll
        for (int i = 0; i < 4; ++i) {
            ra[i] = make_float4(0.f, 0.f, 0.f, 0.f);
            if (k0 + ec[i] < K)
                ra[i] = *(const float4*)(A + (long)(mBase + er[i]) * lda + (k0 + ec[i]));
            rw[i] = make_float4(0.f, 0.f, 0.f, 0.f);
            if ((nBase + er[i]) < N && (k0 + ec[i]) < K)
                rw[i] = *(const float4*)(W + (long)(nBase + er[i]) * K + (k0 + ec[i]));
        }
    };
    auto lstore = [&](int buf) {
        #pragma unroll
        for (int i = 0; i < 4; ++i) {
            *(v4bf*)&As[buf][er[i]][ec[i]] = cvt4(ra[i]);   // 2x v_cvt_pk + ds_store_b64
            *(v4bf*)&Ws[buf][er[i]][ec[i]] = cvt4(rw[i]);
        }
    };

    v8f acc[2][4] = {};

    const int NK = (K + KT - 1) / KT;
    gload(0);
    lstore(0);
    __syncthreads();

    for (int kt = 0; kt < NK; ++kt) {
        const int cur = kt & 1;
        if (kt + 1 < NK) gload((kt + 1) * KT);   // overlap HBM with WMMA

        v16bf a[2], b[4];
        #pragma unroll
        for (int mi = 0; mi < 2; ++mi)
            a[mi] = *(const v16bf*)&As[cur][wm * 32 + mi * 16 + row16][kh];
        #pragma unroll
        for (int ni = 0; ni < 4; ++ni)
            b[ni] = *(const v16bf*)&Ws[cur][wn * 64 + ni * 16 + row16][kh];

        #pragma unroll
        for (int mi = 0; mi < 2; ++mi)
            #pragma unroll
            for (int ni = 0; ni < 4; ++ni)
                acc[mi][ni] = __builtin_amdgcn_wmma_f32_16x16x32_bf16(
                    false, a[mi], false, b[ni], (short)0, acc[mi][ni],
                    false, false);

        if (kt + 1 < NK) lstore(cur ^ 1);        // other buffer: no barrier needed
        __syncthreads();
    }

    // ---- store: 16x16 f32 C/D layout: VGPR r -> row r + 8*(lane>>4), col lane&15
    const int rhalf = (lane >> 4) << 3;  // 0 or 8
    #pragma unroll
    for (int mi = 0; mi < 2; ++mi) {
        #pragma unroll
        for (int ni = 0; ni < 4; ++ni) {
            int col   = nBase + wn * 64 + ni * 16 + row16;
            int rbase = mBase + wm * 32 + mi * 16 + rhalf;
            if (col < N) {
                #pragma unroll
                for (int r = 0; r < 8; ++r) {
                    float v = acc[mi][ni][r];
                    if constexpr (EPILOGUE == 1) {
                        v += bias[col];
                        v = (v > 20.f) ? v : log1pf(__expf(v));   // softplus
                    }
                    C[(long)(rbase + r) * ldc + col] = v;
                }
            }
        }
    }
}

// ---------------------------------------------------------------------------
// Depthwise causal conv1d (k=4, left pad 3) + bias + SiLU.
// ---------------------------------------------------------------------------
__global__ __launch_bounds__(256)
void conv_silu_kernel(const float* __restrict__ xr,
                      const float* __restrict__ cw,   // [DINNER, 1, DCONV]
                      const float* __restrict__ cb,   // [DINNER]
                      float* __restrict__ u)
{
    long idx = (long)blockIdx.x * blockDim.x + threadIdx.x;
    const long total = (long)BSZ * LSEQ * DINNER;
    if (idx >= total) return;
    int  d  = (int)(idx % DINNER);
    long bl = idx / DINNER;
    int  l  = (int)(bl % LSEQ);
    long b  = bl / LSEQ;

    const float* w = cw + (long)d * DCONV;
    float acc = cb[d];
    #pragma unroll
    for (int j = 0; j < DCONV; ++j) {
        int li = l - (DCONV - 1) + j;
        if (li >= 0)
            acc += w[j] * xr[((b * LSEQ) + li) * (long)(2 * DINNER) + d];
    }
    u[idx] = acc / (1.f + __expf(-acc));   // SiLU
}

// ---------------------------------------------------------------------------
// Selective scan. One thread per (b, d) channel; h[16] lives in VGPRs.
// B_t/C_t chunks (16 steps x 32 f32, row stride 80 f32 in xp) are DMA'd
// global->LDS by the Tensor Data Mover (wave 0 issues one tensor_load_to_lds
// per chunk; EXEC is ignored by TDM, completion via s_wait_tensorcnt), with
// a cooperative-load fallback when the toolchain lacks the TDM builtins.
// Fused epilogue: y = (scan + u*D) * silu(res).
// ---------------------------------------------------------------------------
#define SCAN_TPB 256
#define SCAN_CHUNK 16

__global__ __launch_bounds__(SCAN_TPB)
void scan_kernel(const float* __restrict__ delta,  // [M, DINNER]
                 const float* __restrict__ u,      // [M, DINNER]
                 const float* __restrict__ xp,     // [M, 80] (B at 48, C at 64)
                 const float* __restrict__ xr,     // [M, 3072] (res at 1536)
                 const float* __restrict__ A_log,  // [DINNER, DSTATE]
                 const float* __restrict__ Dp,     // [DINNER]
                 float* __restrict__ y)            // [M, DINNER]
{
    const int d = blockIdx.x * SCAN_TPB + threadIdx.x;
    const int b = blockIdx.y;

    // BC[t][0..15] = B_t, BC[t][16..31] = C_t
    __shared__ __align__(16) float BC[SCAN_CHUNK][2 * DSTATE];

    float Arow[DSTATE];
    #pragma unroll
    for (int n = 0; n < DSTATE; ++n)
        Arow[n] = -__expf(A_log[(long)d * DSTATE + n]);
    const float Dd = Dp[d];

    float h[DSTATE];
    #pragma unroll
    for (int n = 0; n < DSTATE; ++n) h[n] = 0.f;

    const long base_di = (long)b * LSEQ * DINNER;
    const long base_xp = (long)b * LSEQ * XPN;
    const long base_xr = (long)b * LSEQ * (2 * DINNER);

    for (int l0 = 0; l0 < LSEQ; l0 += SCAN_CHUNK) {
        __syncthreads();   // previous chunk's LDS reads complete
#ifdef USE_TDM
        if (threadIdx.x < 32) {   // wave 0 issues the TDM descriptor
            unsigned long long ga =
                (unsigned long long)(const void*)(xp + base_xp + (long)l0 * XPN + DTRANK);
            unsigned lds = (unsigned)(unsigned long long)(const void*)&BC[0][0];
            // D# group 0: count=1 | lds_addr | global_addr[56:0] | type=2
            u32x4 g0;
            g0[0] = 1u;
            g0[1] = lds;
            g0[2] = (unsigned)ga;
            g0[3] = ((unsigned)(ga >> 32) & 0x01FFFFFFu) | (2u << 30);
            // D# group 1: data_size=4B; tensor 32 x 16 (stride 80); tile 32 x 16
            u32x8 g1;
            g1[0] = (2u << 16);                   // data_size = 4 bytes
            g1[1] = (32u << 16);                  // tensor_dim0 = 32 (lo16)
            g1[2] = (16u << 16);                  // dim0 hi = 0 | tensor_dim1 = 16
            g1[3] = (32u << 16);                  // dim1 hi = 0 | tile_dim0 = 32
            g1[4] = 16u;                          // tile_dim1 = 16, tile_dim2 = 0
            g1[5] = 80u;                          // tensor_dim0_stride = 80 (lo32)
            g1[6] = 0u;                           // stride0 hi | stride1 lo
            g1[7] = 0u;
            asm volatile("tensor_load_to_lds %0, %1" :: "s"(g0), "s"(g1) : "memory");
            __builtin_amdgcn_s_wait_tensorcnt(0);
        }
#else
        for (int i = threadIdx.x; i < SCAN_CHUNK * 2 * DSTATE; i += SCAN_TPB) {
            int t = i >> 5;
            int r = i & 31;
            BC[t][r] = xp[base_xp + (long)(l0 + t) * XPN + DTRANK + r];
        }
#endif
        __syncthreads();

        for (int t = 0; t < SCAN_CHUNK; ++t) {
            const long off = base_di + (long)(l0 + t) * DINNER + d;
            const float dlt = delta[off];
            const float ut  = u[off];
            const float du  = dlt * ut;
            float yt = 0.f;
            #pragma unroll
            for (int n = 0; n < DSTATE; ++n) {
                float dA = __expf(dlt * Arow[n]);
                h[n] = h[n] * dA + du * BC[t][n];
                yt  += h[n] * BC[t][DSTATE + n];
            }
            yt += ut * Dd;
            const float r = xr[base_xr + (long)(l0 + t) * (2 * DINNER) + DINNER + d];
            y[off] = yt * (r / (1.f + __expf(-r)));   // * silu(res)
        }
    }
}

// ---------------------------------------------------------------------------
extern "C" void kernel_launch(void* const* d_in, const int* in_sizes, int n_in,
                              void* d_out, int out_size, void* d_ws, size_t ws_size,
                              hipStream_t stream)
{
    const float* x          = (const float*)d_in[0];
    const float* in_proj_w  = (const float*)d_in[1];
    const float* conv_w     = (const float*)d_in[2];
    const float* conv_b     = (const float*)d_in[3];
    const float* x_proj_w   = (const float*)d_in[4];
    const float* dt_proj_w  = (const float*)d_in[5];
    const float* dt_proj_b  = (const float*)d_in[6];
    const float* A_log      = (const float*)d_in[7];
    const float* Dp         = (const float*)d_in[8];
    const float* out_proj_w = (const float*)d_in[9];
    float* out = (float*)d_out;

    const long M = (long)BSZ * LSEQ;   // 32768

    // workspace layout (f32):
    char* ws = (char*)d_ws;
    float* xr    = (float*)ws;  ws += (size_t)M * (2 * DINNER) * sizeof(float); // 403 MB
    float* u     = (float*)ws;  ws += (size_t)M * DINNER       * sizeof(float); // 201 MB
    float* xp    = (float*)ws;  ws += (size_t)M * XPN          * sizeof(float); //  11 MB
    float* delta = (float*)ws;  ws += (size_t)M * DINNER       * sizeof(float); // 201 MB
    float* y     = (float*)ws;                                                  // 201 MB

    dim3 blk(256);

    // 1. in_proj: xr = x @ in_proj_w^T        [32768 x 3072], K=768
    gemm_bf16_wmma<0><<<dim3((2 * DINNER + 127) / 128, M / 128), blk, 0, stream>>>(
        x, DMODEL, in_proj_w, nullptr, xr, 2 * DINNER, (int)M, 2 * DINNER, DMODEL);

    // 2. causal depthwise conv + SiLU -> u    [32768 x 1536]
    const long tot = M * DINNER;
    conv_silu_kernel<<<dim3((unsigned)((tot + 255) / 256)), blk, 0, stream>>>(
        xr, conv_w, conv_b, u);

    // 3. x_proj: xp = u @ x_proj_w^T          [32768 x 80], K=1536
    gemm_bf16_wmma<0><<<dim3((XPN + 127) / 128, M / 128), blk, 0, stream>>>(
        u, DINNER, x_proj_w, nullptr, xp, XPN, (int)M, XPN, DINNER);

    // 4. dt_proj + softplus: delta = softplus(xp[:, :48] @ dt_proj_w^T + b)
    gemm_bf16_wmma<1><<<dim3((DINNER + 127) / 128, M / 128), blk, 0, stream>>>(
        xp, XPN, dt_proj_w, dt_proj_b, delta, DINNER, (int)M, DINNER, DTRANK);

    // 5. selective scan + gating -> y         [32768 x 1536]
    scan_kernel<<<dim3(DINNER / SCAN_TPB, BSZ), dim3(SCAN_TPB), 0, stream>>>(
        delta, u, xp, xr, A_log, Dp, y);

    // 6. out_proj: out = y @ out_proj_w^T     [32768 x 768], K=1536
    gemm_bf16_wmma<0><<<dim3((DMODEL + 127) / 128, M / 128), blk, 0, stream>>>(
        y, DINNER, out_proj_w, nullptr, out, DMODEL, (int)M, DMODEL, DINNER);
}
